// ModeloNeuralVasicek_88158498717914
// MI455X (gfx1250) — matrix operand order repaired and checked by reading, not compiled, not run on hardware
//
#include <hip/hip_runtime.h>
#include <math.h>

// ---------------------------------------------------------------------------
// ModeloNeuralVasicek on MI455X (gfx1250):
//  - single persistent 1024-thread workgroup (32 wave32 waves on one WGP)
//  - ctx @ W1 hoisted out of the 2520-step loop (time-invariant) -> C0 in regs
//  - per-step layer-2 GEMM via v_wmma_f32_16x16x32_f16 (weights in LDS,
//    packed in exact CDNA5 B-fragment layout; h1 in LDS A-friendly layout)
//  - layer-3 (128->1) fused into the WMMA epilogue: gelu * W3[N], butterfly
//    __shfl_xor reduction, ds_add_f32 accumulation
//  - double-buffered rmean/acc: 3 barriers per step instead of 4
//  - branch-free GELU/softplus (v_exp_f32 + v_rcp_f32, no exec-mask branches)
//  - path state (r, area) lives in registers: 16 paths/thread, 64x256 total
// ---------------------------------------------------------------------------

typedef _Float16 h16;
typedef __attribute__((ext_vector_type(8)))  _Float16 v8h;
typedef __attribute__((ext_vector_type(16))) _Float16 v16h;
typedef __attribute__((ext_vector_type(8)))  float    v8f;

namespace {
constexpr float DTf  = 1.0f / 252.0f;
constexpr float SQDT = 0.06299407883487120f;   // sqrt(1/252)

// ---- LDS layout (bytes), total = 125 KB -----------------------------------
constexpr int OFF_W2T   = 0;       // 256 cols x 136 h16 (k-major, padded) = 69632
constexpr int OFF_H1    = 69632;   // 64 rows x 264 h16 (row-major, padded) = 33792
constexpr int OFF_CTX   = 103424;  // 64 x 192 h16 (phase A only)           = 24576
// loop-time smalls live inside the dead ctx region:
constexpr int OFF_WR    = 103424;  // 256 h16  (W1 row 0: r_mean weights)
constexpr int OFF_WT    = 103936;  // 256 h16  (W1 row 1: t weights)
constexpr int OFF_B2    = 104448;  // 256 h16  (layer-2 bias, cat)
constexpr int OFF_W3    = 104960;  // 256 h16  (layer-3 weights, cat)
constexpr int OFF_RMEAN = 105472;  // 2 x 64 f32  (double-buffered)
constexpr int OFF_ACC   = 105984;  // 2 x 128 f32 (double-buffered: mu | si)
constexpr int OFF_MISC  = 107008;  // f32: b3_mu, b3_si
constexpr int SMEM_SZ   = 128000;  // 125.0 KB
// phase-A aliases (dead before the loop starts)
constexpr int OFF_MLN   = 0;              // 64x256 h16 LN means   (inside W2T)
constexpr int OFF_ILN   = 32768;          // 64x256 h16 LN inv-std (inside W2T)
constexpr int OFF_SUM   = OFF_H1;         // 64x64 f32 sum over T  (inside H1)
constexpr int OFF_SSQ   = OFF_H1 + 16384; // 64x64 f32 sumsq       (inside H1)
} // namespace

// Branch-free GELU: x * sigmoid(1.5957691*(x + 0.044715*x^3))
// (tanh-form; v_exp_f32 + v_rcp_f32, no exec-mask divergence)
__device__ __forceinline__ float gelu_f(float x) {
  float z = 1.59576912160573f * fmaf(0.044715f * x * x, x, x);
  return x * __builtin_amdgcn_rcpf(1.0f + __expf(-z));
}

// Branch-free softplus: max(x,0) + log(1 + exp(-|x|))
__device__ __forceinline__ float softplus_f(float x) {
  return fmaxf(x, 0.0f) + __logf(1.0f + __expf(-fabsf(x)));
}

__device__ __forceinline__ unsigned pcg32(unsigned& s) {
  s = s * 747796405u + 2891336453u;
  unsigned w = ((s >> ((s >> 28) + 4u)) ^ s) * 277803737u;
  return (w >> 22) ^ w;
}

__global__ __launch_bounds__(1024, 1)
void vasicek_sde_kernel(const float* __restrict__ X,    const float* __restrict__ rU,
                        const float* __restrict__ mat,  const float* __restrict__ Wp,
                        const float* __restrict__ bp,   const float* __restrict__ lng,
                        const float* __restrict__ lnb,
                        const float* __restrict__ muW1, const float* __restrict__ mub1,
                        const float* __restrict__ muW2, const float* __restrict__ mub2,
                        const float* __restrict__ muW3, const float* __restrict__ mub3,
                        const float* __restrict__ siW1, const float* __restrict__ sib1,
                        const float* __restrict__ siW2, const float* __restrict__ sib2,
                        const float* __restrict__ siW3, const float* __restrict__ sib3,
                        float* __restrict__ out, int nmat) {
  __shared__ __align__(16) char sm[SMEM_SZ];

  const int tid  = threadIdx.x;
  const int lane = tid & 31;
  const int wave = tid >> 5;
  const int b    = tid >> 4;          // batch row owned by this thread (16 thr/row)
  const int n0   = (tid & 15) * 16;   // 16 contiguous cat-columns owned (phase 2)

  // n_steps from maturities (device-side, capture-safe)
  float maxT = 0.0f;
  for (int m = 0; m < nmat; ++m) maxT = fmaxf(maxT, mat[m]);
  int n_steps = (int)ceilf(maxT * 252.0f);
  if (n_steps < 2) n_steps = 2;

  // ========================================================================
  // Phase A1: per-(b,t) LayerNorm statistics of tanh(X @ Wp + bp)
  // ========================================================================
  {
    h16* mLN = (h16*)(sm + OFF_MLN);
    h16* iLN = (h16*)(sm + OFF_ILN);
    for (int it = 0; it < 16; ++it) {
      int p = it * 1024 + tid;
      int bb = p >> 8, tt = p & 255;
      float x0 = X[(bb * 256 + tt) * 2 + 0];
      float x1 = X[(bb * 256 + tt) * 2 + 1];
      float s = 0.0f, s2 = 0.0f;
      for (int j = 0; j < 64; ++j) {
        float hv = tanhf(x0 * Wp[j] + x1 * Wp[64 + j] + bp[j]);
        s += hv; s2 += hv * hv;
      }
      float mm = s * (1.0f / 64.0f);
      float vv = s2 * (1.0f / 64.0f) - mm * mm;
      mLN[bb * 256 + tt] = (h16)mm;
      iLN[bb * 256 + tt] = (h16)rsqrtf(vv + 1e-5f);
    }
  }
  __syncthreads();

  // ========================================================================
  // Phase A2: per-(b,j) accumulate sum/sumsq over T of normalized H; last row
  // ========================================================================
  {
    const h16* mLN = (const h16*)(sm + OFF_MLN);
    const h16* iLN = (const h16*)(sm + OFF_ILN);
    float* sumT = (float*)(sm + OFF_SUM);
    float* ssqT = (float*)(sm + OFF_SSQ);
    h16*   ctx  = (h16*)(sm + OFF_CTX);
    for (int it = 0; it < 4; ++it) {
      int p = it * 1024 + tid;
      int bb = p >> 6, j = p & 63;
      float w0 = Wp[j], w1 = Wp[64 + j], b0 = bp[j];
      float g = lng[j], be = lnb[j];
      float s = 0.0f, s2 = 0.0f, lastg = 0.0f;
      for (int tt = 0; tt < 256; ++tt) {
        float x0 = X[(bb * 256 + tt) * 2 + 0];
        float x1 = X[(bb * 256 + tt) * 2 + 1];
        float hv = tanhf(x0 * w0 + x1 * w1 + b0);
        float gg = (hv - (float)mLN[bb * 256 + tt]) * (float)iLN[bb * 256 + tt] * g + be;
        s += gg; s2 += gg * gg; lastg = gg;
      }
      sumT[bb * 64 + j] = s;
      ssqT[bb * 64 + j] = s2;
      ctx[bb * 192 + 128 + j] = (h16)lastg;   // H[:, -1, :]
    }
  }
  __syncthreads();

  // ========================================================================
  // Phase A3: ctx = [mean_T, std_T(ddof=1), last]
  // ========================================================================
  {
    const float* sumT = (const float*)(sm + OFF_SUM);
    const float* ssqT = (const float*)(sm + OFF_SSQ);
    h16* ctx = (h16*)(sm + OFF_CTX);
    for (int it = 0; it < 4; ++it) {
      int p = it * 1024 + tid;
      int bb = p >> 6, j = p & 63;
      float s = sumT[bb * 64 + j], s2 = ssqT[bb * 64 + j];
      float mn = s * (1.0f / 256.0f);
      float var = (s2 - 256.0f * mn * mn) * (1.0f / 255.0f);
      ctx[bb * 192 + j]      = (h16)mn;
      ctx[bb * 192 + 64 + j] = (h16)sqrtf(fmaxf(var, 0.0f));
    }
  }
  __syncthreads();

  // ========================================================================
  // Phase A4: C0 = ctx @ W1[2:,:] + b1  (time-invariant!) -> registers
  //           (done BEFORE constant packing so ctx region can be reused)
  // ========================================================================
  float c0r[16];
  {
    const h16* ctx = (const h16*)(sm + OFF_CTX);
    const float* b1p = (n0 < 128) ? (mub1 + n0) : (sib1 + (n0 - 128));
    const float* W1p = (n0 < 128) ? muW1 : siW1;
    const int nc = (n0 < 128) ? n0 : (n0 - 128);
#pragma unroll
    for (int i = 0; i < 16; ++i) c0r[i] = b1p[i];
    for (int k = 0; k < 192; ++k) {
      float cv = (float)ctx[b * 192 + k];
      const float* wrow = W1p + (2 + k) * 128 + nc;
#pragma unroll
      for (int i = 0; i < 16; ++i) c0r[i] = fmaf(cv, wrow[i], c0r[i]);
    }
  }
  __syncthreads();   // ctx reads done; region is now reusable

  // ========================================================================
  // Phase A5: pack W2 (cat, f16) into CDNA5 B-fragment layout (column-major,
  //           K contiguous per column, stride 136); pack loop-time constants
  //           into the dead ctx region; init double buffers.
  // ========================================================================
  {
    h16* W2T = (h16*)(sm + OFF_W2T);
    for (int it = 0; it < 32; ++it) {
      int idx = it * 1024 + tid;
      int n = idx >> 7, k = idx & 127;
      float w = (n < 128) ? muW2[k * 128 + n] : siW2[k * 128 + (n - 128)];
      W2T[n * 136 + k] = (h16)w;
    }
    if (tid < 256) {
      int n = tid;
      h16* wr = (h16*)(sm + OFF_WR); h16* wt = (h16*)(sm + OFF_WT);
      h16* b2 = (h16*)(sm + OFF_B2); h16* w3 = (h16*)(sm + OFF_W3);
      wr[n] = (h16)((n < 128) ? muW1[n]       : siW1[n - 128]);
      wt[n] = (h16)((n < 128) ? muW1[128 + n] : siW1[128 + (n - 128)]);
      b2[n] = (h16)((n < 128) ? mub2[n]       : sib2[n - 128]);
      w3[n] = (h16)((n < 128) ? muW3[n]       : siW3[n - 128]);
      ((float*)(sm + OFF_ACC))[n] = 0.0f;          // zero both acc buffers
    }
    if (tid < 64) ((float*)(sm + OFF_RMEAN))[tid] = rU[tid];  // rmean buf 0
    if (tid == 0) {
      float* misc = (float*)(sm + OFF_MISC);
      misc[0] = mub3[0]; misc[1] = sib3[0];
    }
  }

  // persistent path state: 16 paths per thread (64 rows x 256 paths)
  float rp[16], ar[16];
  {
    float r0 = rU[b];
#pragma unroll
    for (int j = 0; j < 16; ++j) { rp[j] = r0; ar[j] = 0.0f; }
  }
  unsigned rs = 0x9E3779B9u * (unsigned)(tid + 1) + 0x85EBCA6Bu;
  (void)pcg32(rs); (void)pcg32(rs);

  __syncthreads();
  const float b3m = ((const float*)(sm + OFF_MISC))[0];
  const float b3s = ((const float*)(sm + OFF_MISC))[1];

  float* rmean = (float*)(sm + OFF_RMEAN);   // [2][64]
  float* acc   = (float*)(sm + OFF_ACC);     // [2][128]  (mu: 0..63, si: 64..127)
  h16*   h1    = (h16*)(sm + OFF_H1);
  h16*   W2T   = (h16*)(sm + OFF_W2T);
  const h16* wrh = (const h16*)(sm + OFF_WR);
  const h16* wth = (const h16*)(sm + OFF_WT);
  const h16* b2h = (const h16*)(sm + OFF_B2);
  const h16* w3h = (const h16*)(sm + OFF_W3);

  const float inv_nm1 = 1.0f / (float)(n_steps - 1);

  // ========================================================================
  // Main Euler-Maruyama loop: 3 barriers/step, double-buffered rmean/acc
  // ========================================================================
  for (int step = 0; step < n_steps; ++step) {
    const int p = step & 1, q = p ^ 1;
    const float tcur = (float)step * inv_nm1;

    // --- phase 2: layer-1 = C0 + r_mean*w_r + t*w_t, GELU -> h1 (f16) -------
    {
      float rm = rmean[p * 64 + b];
      v8h lo, hi;
#pragma unroll
      for (int i = 0; i < 8; ++i) {
        int n = n0 + i;
        float pre = c0r[i] + rm * (float)wrh[n] + tcur * (float)wth[n];
        lo[i] = (h16)gelu_f(pre);
      }
#pragma unroll
      for (int i = 0; i < 8; ++i) {
        int n = n0 + 8 + i;
        float pre = c0r[8 + i] + rm * (float)wrh[n] + tcur * (float)wth[n];
        hi[i] = (h16)gelu_f(pre);
      }
      *(v8h*)(h1 + b * 264 + n0)     = lo;
      *(v8h*)(h1 + b * 264 + n0 + 8) = hi;
    }
    __syncthreads();

    // --- phase 3: layer-2 GEMM (WMMA f16) + fused gelu*W3 epilogue ----------
#pragma unroll
    for (int tt = 0; tt < 2; ++tt) {
      const int tile  = wave * 2 + tt;
      const int mtile = tile >> 4;
      const int nt    = tile & 15;
      const int rowM  = mtile * 16 + (lane & 15);
      const int colN  = nt * 16 + (lane & 15);
      const int kbase = (nt < 8) ? 0 : 128;       // mu-half vs si-half of h1
      const int kaOff = (lane >= 16) ? 8 : 0;     // A-fragment lane K offset
      const int kbOff = (lane >= 16) ? 8 : 0;     // B-fragment lane K offset
      const h16* arow = h1 + rowM * 264 + kbase + kaOff;
      const h16* bcol = W2T + colN * 136 + kbOff;

      v8f c = {0.f, 0.f, 0.f, 0.f, 0.f, 0.f, 0.f, 0.f};
#pragma unroll
      for (int kc = 0; kc < 4; ++kc) {
        v8h alo = *(const v8h*)(arow + kc * 32);
        v8h ahi = *(const v8h*)(arow + kc * 32 + 16);
        v16h A = __builtin_shufflevector(alo, ahi, 0, 1, 2, 3, 4, 5, 6, 7,
                                                   8, 9, 10, 11, 12, 13, 14, 15);
        v8h blo = *(const v8h*)(bcol + kc * 32);
        v8h bhi = *(const v8h*)(bcol + kc * 32 + 16);
        v16h Bm = __builtin_shufflevector(blo, bhi, 0, 1, 2, 3, 4, 5, 6, 7,
                                                    8, 9, 10, 11, 12, 13, 14, 15);
        c = __builtin_amdgcn_wmma_f32_16x16x32_f16(false, A, false, Bm,
                                                   (short)0, c, false, false);
      }

      // epilogue: h2 = gelu(c + b2), partial dot with W3 column weight
      const float b2v = (float)b2h[colN];
      const float w3v = (float)w3h[colN];
      float pr[8];
#pragma unroll
      for (int r = 0; r < 8; ++r) pr[r] = gelu_f(c[r] + b2v) * w3v;
      // reduce over the tile's 16 columns (lanes 0-15 and 16-31 independently)
#pragma unroll
      for (int r = 0; r < 8; ++r) {
        pr[r] += __shfl_xor(pr[r], 1, 32);
        pr[r] += __shfl_xor(pr[r], 2, 32);
        pr[r] += __shfl_xor(pr[r], 4, 32);
        pr[r] += __shfl_xor(pr[r], 8, 32);
      }
      if ((lane & 15) == 0) {
        const int rbase = mtile * 16 + ((lane >= 16) ? 8 : 0);
        float* dst = acc + p * 128 + ((nt < 8) ? 0 : 64);
#pragma unroll
        for (int r = 0; r < 8; ++r) atomicAdd(&dst[rbase + r], pr[r]);
      }
    }
    __syncthreads();

    // --- phase 4: mu / softplus(si), Box-Muller RNG, path + area update; ----
    // --- then immediately produce next step's r-mean + reset other buffer ---
    {
      float mu = acc[p * 128 + b] + b3m;
      float sv = acc[p * 128 + 64 + b] + b3s;
      float si = softplus_f(sv) + 1e-5f;
      const float drift = mu * DTf;
      const float diff  = si * SQDT;
#pragma unroll
      for (int j = 0; j < 16; j += 2) {
        float u1 = (float)(pcg32(rs) >> 8) * (1.0f / 16777216.0f) + 5.9604645e-8f;
        float u2 = (float)(pcg32(rs) >> 8) * (1.0f / 16777216.0f);
        float rad = sqrtf(-2.0f * __logf(u1));
        float sn, cs;
        __sincosf(6.28318530717958647f * u2, &sn, &cs);
        rp[j]     += drift + diff * (rad * cs);
        ar[j]     += rp[j] * DTf;
        rp[j + 1] += drift + diff * (rad * sn);
        ar[j + 1] += rp[j + 1] * DTf;
      }
      float s = 0.0f;
#pragma unroll
      for (int j = 0; j < 16; ++j) s += rp[j];
      s += __shfl_xor(s, 1, 32); s += __shfl_xor(s, 2, 32);
      s += __shfl_xor(s, 4, 32); s += __shfl_xor(s, 8, 32);
      if ((lane & 15) == 0) {            // lanes 0 / 16 own rows 2w / 2w+1
        rmean[q * 64 + b] = s * (1.0f / 256.0f);
        acc[q * 128 + b] = 0.0f;
        acc[q * 128 + 64 + b] = 0.0f;
      }
    }
    __syncthreads();
  }

  // ========================================================================
  // Epilogue: area mean per row -> integrais (B, nmat)
  // ========================================================================
  {
    float s = 0.0f;
#pragma unroll
    for (int j = 0; j < 16; ++j) s += ar[j];
    s += __shfl_xor(s, 1, 32); s += __shfl_xor(s, 2, 32);
    s += __shfl_xor(s, 4, 32); s += __shfl_xor(s, 8, 32);
    if ((lane & 15) == 0) rmean[b] = s;   // reuse buf0 as area-sum
  }
  __syncthreads();
  if (tid < 64 * nmat) {
    int bb = tid / nmat, mm = tid - bb * nmat;
    float m = mat[mm];
    float frac = m / (maxT + 1e-12f);
    out[bb * nmat + mm] = (rmean[bb] * (1.0f / 256.0f)) * frac / (m + 1e-12f);
  }
}

extern "C" void kernel_launch(void* const* d_in, const int* in_sizes, int n_in,
                              void* d_out, int out_size, void* d_ws, size_t ws_size,
                              hipStream_t stream) {
  const float* X    = (const float*)d_in[0];
  const float* rU   = (const float*)d_in[1];
  const float* mat  = (const float*)d_in[2];
  const float* Wp   = (const float*)d_in[3];
  const float* bp   = (const float*)d_in[4];
  const float* lng  = (const float*)d_in[5];
  const float* lnb  = (const float*)d_in[6];
  const float* muW1 = (const float*)d_in[7];
  const float* mub1 = (const float*)d_in[8];
  const float* muW2 = (const float*)d_in[9];
  const float* mub2 = (const float*)d_in[10];
  const float* muW3 = (const float*)d_in[11];
  const float* mub3 = (const float*)d_in[12];
  const float* siW1 = (const float*)d_in[13];
  const float* sib1 = (const float*)d_in[14];
  const float* siW2 = (const float*)d_in[15];
  const float* sib2 = (const float*)d_in[16];
  const float* siW3 = (const float*)d_in[17];
  const float* sib3 = (const float*)d_in[18];
  float* out = (float*)d_out;
  int nmat = in_sizes[2];

  hipLaunchKernelGGL(vasicek_sde_kernel, dim3(1), dim3(1024), 0, stream,
                     X, rU, mat, Wp, bp, lng, lnb,
                     muW1, mub1, muW2, mub2, muW3, mub3,
                     siW1, sib1, siW2, sib2, siW3, sib3,
                     out, nmat);
}
